// WriteFirstModel_35270271435195
// MI455X (gfx1250) — matrix operand order repaired and checked by reading, not compiled, not run on hardware
//
#include <hip/hip_runtime.h>
#include <hip/hip_bf16.h>
#include <math.h>

typedef __bf16 bf16;
typedef bf16  v16bf  __attribute__((ext_vector_type(16)));
typedef bf16  bf16x8 __attribute__((ext_vector_type(8)));
typedef float v8f    __attribute__((ext_vector_type(8)));

#define D 512
#define TT 2048
#define BB 32
#define VOCAB 50257
#define ROWS (BB * TT)          // 65536
#define HID  (2 * D)            // 1024

// ---------------------------------------------------------------------------
// prep: convert W1/W2 to bf16, compute wg_mean = mean over slot columns of Wg
// ---------------------------------------------------------------------------
__global__ __launch_bounds__(256) void prep_kernel(
    const float* __restrict__ W1, const float* __restrict__ W2,
    const float* __restrict__ Wg,
    bf16* __restrict__ w1b, bf16* __restrict__ w2b, float* __restrict__ wgm)
{
    long i = (long)blockIdx.x * 256 + threadIdx.x;
    if (i < (long)D * HID) { w1b[i] = (bf16)W1[i]; w2b[i] = (bf16)W2[i]; }
    if (i < D) {
        float s = 0.f;
        #pragma unroll
        for (int j = 0; j < 16; ++j) s += Wg[i * 16 + j];
        wgm[i] = s * (1.f / 16.f);
    }
}

// ---------------------------------------------------------------------------
// gather: h = embed[seq]  (f32 copy for residual, bf16 copy for WMMA A)
// ---------------------------------------------------------------------------
__global__ __launch_bounds__(256) void gather_kernel(
    const int* __restrict__ seq, const float* __restrict__ embed,
    float* __restrict__ hf, bf16* __restrict__ hb)
{
    long i   = (long)blockIdx.x * 256 + threadIdx.x;   // over ROWS*D
    long row = i >> 9;
    int  d   = (int)(i & 511);
    int  tok = seq[row];
    float v  = embed[(long)tok * D + d];
    hf[i] = v;
    hb[i] = (bf16)v;
}

// ---------------------------------------------------------------------------
// bf16 WMMA GEMM: C[M,N] = A[M,K] * B[K,N]
//   tile 128(M) x 64(N), K stepped by 64, double-buffered LDS ping-pong
//   (one barrier per iteration, next tile's global loads overlap WMMAs).
//   8 waves, each owns a 32x32 patch -> 8 v_wmma per iteration.
//   mode 0: out_b = bf16(relu(acc + bias))
//   mode 1: out_f = acc + bias + resid
// ---------------------------------------------------------------------------
__global__ __launch_bounds__(256) void ffgemm_kernel(
    const bf16* __restrict__ A, int lda,
    const bf16* __restrict__ B, int ldb,
    const float* __restrict__ bias,
    const float* __restrict__ resid,
    bf16* __restrict__ out_b, float* __restrict__ out_f,
    int N, int K, int mode)
{
    __shared__ __align__(16) bf16 As[2][128][72];   // 64-K chunk, padded stride
    __shared__ __align__(16) bf16 Bs[2][64][72];    // transposed: [n][k]

    const int tid  = threadIdx.x;
    const int wave = tid >> 5;
    const int lane = tid & 31;
    const int half = lane >> 4;
    const int lr   = lane & 15;
    const int wm   = wave & 3;        // 4 waves along M
    const int wn   = wave >> 2;       // 2 waves along N
    const long m_base = (long)blockIdx.x * 128;
    const int  n_base = blockIdx.y * 64;

    v8f zero = {};
    v8f acc[2][2];
    acc[0][0] = zero; acc[0][1] = zero; acc[1][0] = zero; acc[1][1] = zero;

    bf16x8 aReg[4];   // A tile: 128x64 = 1024 chunks of 8, 4 per thread
    bf16x8 bReg[2];   // B tile:  64x64 =  512 chunks of 8, 2 per thread

    auto gload = [&](int k0) {
        #pragma unroll
        for (int i = 0; i < 4; ++i) {
            int c = tid + i * 256;
            int r = c >> 3, cc = (c & 7) * 8;
            aReg[i] = *(const bf16x8*)(A + (m_base + r) * (long)lda + k0 + cc);
        }
        #pragma unroll
        for (int i = 0; i < 2; ++i) {
            int c = tid + i * 256;
            int k = c >> 3, nn = (c & 7) * 8;
            bReg[i] = *(const bf16x8*)(B + (long)(k0 + k) * ldb + n_base + nn);
        }
    };
    auto lstore = [&](int buf) {
        #pragma unroll
        for (int i = 0; i < 4; ++i) {
            int c = tid + i * 256;
            int r = c >> 3, cc = (c & 7) * 8;
            *(bf16x8*)(&As[buf][r][cc]) = aReg[i];
        }
        #pragma unroll
        for (int i = 0; i < 2; ++i) {
            int c = tid + i * 256;
            int k = c >> 3, nn = (c & 7) * 8;
            #pragma unroll
            for (int j = 0; j < 8; ++j) Bs[buf][nn + j][k] = bReg[i][j];
        }
    };

    const int nIter = K >> 6;
    gload(0);
    lstore(0);
    __syncthreads();

    for (int it = 0; it < nIter; ++it) {
        const int cur = it & 1;
        if (it + 1 < nIter) {
            gload((it + 1) * 64);
            if (it + 2 < nIter) {   // warm L2 two tiles ahead
                __builtin_prefetch(A + (m_base + (tid >> 1)) * (long)lda
                                     + (it + 2) * 64 + (tid & 1) * 32, 0, 1);
                __builtin_prefetch(B + (long)((it + 2) * 64 + (tid >> 3)) * ldb
                                     + n_base + (tid & 7) * 8, 0, 1);
            }
        }
        // fragments per ISA layout:
        //  A lane(l): row=l&15, K = {8h..8h+7} U {16+8h..16+8h+7} (+32*ks)
        //  B lane(l): col=l&15, K = {8h..8h+7} U {16+8h..16+8h+7} (+32*ks)
        #pragma unroll
        for (int ks = 0; ks < 2; ++ks) {
            v16bf afrag[2], bfrag[2];
            #pragma unroll
            for (int mi = 0; mi < 2; ++mi) {
                int m = wm * 32 + mi * 16 + lr;
                const bf16* base = &As[cur][m][ks * 32];
                bf16x8 lo = *(const bf16x8*)(base + 8 * half);
                bf16x8 hi = *(const bf16x8*)(base + 16 + 8 * half);
                #pragma unroll
                for (int e = 0; e < 8; ++e) { afrag[mi][e] = lo[e]; afrag[mi][e + 8] = hi[e]; }
            }
            #pragma unroll
            for (int ni = 0; ni < 2; ++ni) {
                int n = wn * 32 + ni * 16 + lr;
                const bf16* base = &Bs[cur][n][ks * 32];
                bf16x8 lo = *(const bf16x8*)(base + 8 * half);
                bf16x8 hi = *(const bf16x8*)(base + 16 + 8 * half);
                #pragma unroll
                for (int e = 0; e < 8; ++e) { bfrag[ni][e] = lo[e]; bfrag[ni][e + 8] = hi[e]; }
            }
            #pragma unroll
            for (int mi = 0; mi < 2; ++mi)
                #pragma unroll
                for (int ni = 0; ni < 2; ++ni)
                    acc[mi][ni] = __builtin_amdgcn_wmma_f32_16x16x32_bf16(
                        false, afrag[mi], false, bfrag[ni],
                        (short)0, acc[mi][ni], false, false);
        }
        if (it + 1 < nIter) lstore(cur ^ 1);
        __syncthreads();
    }

    // epilogue; C layout: lane(l) col=l&15, VGPR j -> row = j + 8*(l>>4)
    #pragma unroll
    for (int mi = 0; mi < 2; ++mi) {
        #pragma unroll
        for (int ni = 0; ni < 2; ++ni) {
            int ncol = n_base + wn * 32 + ni * 16 + lr;
            float bv = bias[ncol];
            #pragma unroll
            for (int j = 0; j < 8; ++j) {
                long mrow = m_base + wm * 32 + mi * 16 + j + 8 * half;
                float v = acc[mi][ni][j] + bv;
                if (mode == 0) {
                    v = fmaxf(v, 0.f);
                    out_b[mrow * (long)N + ncol] = (bf16)v;
                } else {
                    out_f[mrow * (long)N + ncol] = v + resid[mrow * (long)N + ncol];
                }
            }
        }
    }
}

// ---------------------------------------------------------------------------
// layernorm in place, one wave32 per row of 512; fused gate score:
// gate[b,t] = dot(LN(row), wg_mean) for t < TT-1 (bg shift irrelevant to topk)
// ---------------------------------------------------------------------------
__global__ __launch_bounds__(256) void ln_kernel(
    float* __restrict__ x, const float* __restrict__ g,
    const float* __restrict__ bta,
    const float* __restrict__ wgm, float* __restrict__ gate, long rows)
{
    long row = (long)blockIdx.x * 8 + (threadIdx.x >> 5);
    int lane = threadIdx.x & 31;
    if (row >= rows) return;
    float* xr = x + row * D;
    float v[16];
    float s = 0.f;
    #pragma unroll
    for (int i = 0; i < 16; ++i) { v[i] = xr[lane + i * 32]; s += v[i]; }
    #pragma unroll
    for (int m = 16; m > 0; m >>= 1) s += __shfl_xor(s, m, 32);
    float mu = s * (1.f / 512.f);
    float s2 = 0.f;
    #pragma unroll
    for (int i = 0; i < 16; ++i) { float d = v[i] - mu; s2 += d * d; }
    #pragma unroll
    for (int m = 16; m > 0; m >>= 1) s2 += __shfl_xor(s2, m, 32);
    float inv = rsqrtf(s2 * (1.f / 512.f) + 1e-5f);
    float gs = 0.f;
    #pragma unroll
    for (int i = 0; i < 16; ++i) {
        int d = lane + i * 32;
        float y = (v[i] - mu) * inv * g[d] + bta[d];
        xr[d] = y;
        gs += y * wgm[d];
    }
    #pragma unroll
    for (int m = 16; m > 0; m >>= 1) gs += __shfl_xor(gs, m, 32);
    long b = row >> 11;           // row / TT
    long t = row & (TT - 1);      // row % TT
    if (lane == 0 && t != TT - 1) gate[b * (TT - 1) + t] = gs;
}

// ---------------------------------------------------------------------------
// top-4 per batch (descending value, ties -> lowest index)
// ---------------------------------------------------------------------------
__global__ __launch_bounds__(256) void topk_kernel(
    const float* __restrict__ gate, int* __restrict__ topidx)
{
    int b = blockIdx.x;
    const float* g = gate + (long)b * (TT - 1);
    __shared__ float sv[256];
    __shared__ int   si[256];
    __shared__ int   sel[4];
    for (int j = 0; j < 4; ++j) {
        float bestv = -INFINITY;
        int   besti = 0x7fffffff;
        for (int t = threadIdx.x; t < TT - 1; t += 256) {
            bool skip = false;
            for (int p = 0; p < j; ++p) if (sel[p] == t) skip = true;
            if (skip) continue;
            float v = g[t];
            if (v > bestv || (v == bestv && t < besti)) { bestv = v; besti = t; }
        }
        sv[threadIdx.x] = bestv; si[threadIdx.x] = besti;
        __syncthreads();
        for (int s = 128; s > 0; s >>= 1) {
            if ((int)threadIdx.x < s) {
                float ov = sv[threadIdx.x + s]; int oi = si[threadIdx.x + s];
                if (ov > sv[threadIdx.x] ||
                    (ov == sv[threadIdx.x] && oi < si[threadIdx.x])) {
                    sv[threadIdx.x] = ov; si[threadIdx.x] = oi;
                }
            }
            __syncthreads();
        }
        if (threadIdx.x == 0) { sel[j] = si[0]; topidx[b * 4 + j] = si[0]; }
        __syncthreads();
    }
}

// ---------------------------------------------------------------------------
// attention: q = h_last@Wr+br; scores over 4 mem rows; softmax incl. 12 zeros
// ---------------------------------------------------------------------------
__global__ __launch_bounds__(256) void attn_kernel(
    const float* __restrict__ hid, const int* __restrict__ topidx,
    const float* __restrict__ Wr, const float* __restrict__ br,
    float* __restrict__ ctx)
{
    int b = blockIdx.x;
    __shared__ float hl[D];
    __shared__ float q[D];
    __shared__ float sc[4];
    __shared__ float w[4];
    const float* hlast = hid + ((long)b * TT + (TT - 1)) * D;
    for (int d = threadIdx.x; d < D; d += 256) hl[d] = hlast[d];
    __syncthreads();
    for (int d = threadIdx.x; d < D; d += 256) {
        float acc = br[d];
        for (int k = 0; k < D; ++k) acc += hl[k] * Wr[(long)k * D + d];
        q[d] = acc;
    }
    __syncthreads();
    int wave = threadIdx.x >> 5, lane = threadIdx.x & 31;
    if (wave < 4) {
        const float* mrow = hid + ((long)b * TT + topidx[b * 4 + wave]) * D;
        float acc = 0.f;
        for (int k = lane; k < D; k += 32) acc += mrow[k] * q[k];
        #pragma unroll
        for (int m = 16; m > 0; m >>= 1) acc += __shfl_xor(acc, m, 32);
        if (lane == 0) sc[wave] = acc;
    }
    __syncthreads();
    if (threadIdx.x == 0) {
        float m = 0.f;   // zero slots participate in the max
        for (int j = 0; j < 4; ++j) m = fmaxf(m, sc[j]);
        float denom = 12.f * expf(-m);     // 12 zero-score slots
        for (int j = 0; j < 4; ++j) denom += expf(sc[j] - m);
        for (int j = 0; j < 4; ++j) w[j] = expf(sc[j] - m) / denom;
    }
    __syncthreads();
    for (int d = threadIdx.x; d < D; d += 256) {
        float acc = 0.f;
        #pragma unroll
        for (int j = 0; j < 4; ++j) {
            const float* mrow = hid + ((long)b * TT + topidx[b * 4 + j]) * D;
            acc += w[j] * mrow[d];
        }
        ctx[(long)b * D + d] = acc;
    }
}

// ---------------------------------------------------------------------------
// output GEMM: each thread owns one vocab column for all 32 batch rows;
// Wo (103 MB) streamed exactly once from HBM.
// ---------------------------------------------------------------------------
__global__ __launch_bounds__(256) void outgemm_kernel(
    const float* __restrict__ ctx, const float* __restrict__ Wo,
    const float* __restrict__ bo, float* __restrict__ out)
{
    __shared__ float cs[BB][128];
    int n = blockIdx.x * 256 + threadIdx.x;
    float acc[BB];
    #pragma unroll
    for (int b = 0; b < BB; ++b) acc[b] = 0.f;
    for (int k0 = 0; k0 < D; k0 += 128) {
        for (int i = threadIdx.x; i < BB * 128; i += 256) {
            int b = i >> 7, k = i & 127;
            cs[b][k] = ctx[(long)b * D + k0 + k];
        }
        __syncthreads();
        if (n < VOCAB) {
            #pragma unroll 4
            for (int k = 0; k < 128; ++k) {
                float wv = Wo[(long)(k0 + k) * VOCAB + n];
                #pragma unroll
                for (int b = 0; b < BB; ++b) acc[b] += cs[b][k] * wv;
            }
        }
        __syncthreads();
    }
    if (n < VOCAB) {
        float bv = bo[n];
        #pragma unroll
        for (int b = 0; b < BB; ++b) out[(long)b * VOCAB + n] = acc[b] + bv;
    }
}

// ---------------------------------------------------------------------------
extern "C" void kernel_launch(void* const* d_in, const int* in_sizes, int n_in,
                              void* d_out, int out_size, void* d_ws, size_t ws_size,
                              hipStream_t stream)
{
    (void)in_sizes; (void)n_in; (void)out_size; (void)ws_size;
    const int*   seq   = (const int*)  d_in[0];
    const float* embed = (const float*)d_in[1];
    const float* W1    = (const float*)d_in[2];
    const float* b1    = (const float*)d_in[3];
    const float* W2    = (const float*)d_in[4];
    const float* b2    = (const float*)d_in[5];
    const float* ln_g  = (const float*)d_in[6];
    const float* ln_b  = (const float*)d_in[7];
    const float* Wg    = (const float*)d_in[8];
    // d_in[9] = bg: uniform shift, irrelevant to top-k ordering
    const float* Wr    = (const float*)d_in[10];
    const float* br    = (const float*)d_in[11];
    const float* Wo    = (const float*)d_in[12];
    const float* bo    = (const float*)d_in[13];
    float* out = (float*)d_out;

    size_t off = 0;
    auto carve = [&](size_t bytes) -> char* {
        off = (off + 255) & ~(size_t)255;
        char* p = (char*)d_ws + off;
        off += bytes;
        return p;
    };
    float* hf    = (float*)carve((size_t)ROWS * D * 4);
    bf16*  hb    = (bf16*) carve((size_t)ROWS * D * 2);
    bf16*  w1b   = (bf16*) carve((size_t)D * HID * 2);
    bf16*  w2b   = (bf16*) carve((size_t)HID * D * 2);
    bf16*  act1  = (bf16*) carve((size_t)ROWS * HID * 2);
    float* hid   = (float*)carve((size_t)ROWS * D * 4);
    float* wgm   = (float*)carve((size_t)D * 4);
    float* gate  = (float*)carve((size_t)BB * (TT - 1) * 4);
    int*   tidx  = (int*)  carve((size_t)BB * 4 * 4);
    float* ctx   = (float*)carve((size_t)BB * D * 4);

    prep_kernel<<<(D * HID + 255) / 256, 256, 0, stream>>>(W1, W2, Wg, w1b, w2b, wgm);
    gather_kernel<<<(ROWS * (long)D) / 256, 256, 0, stream>>>(seq, embed, hf, hb);

    // GEMM1: act1 = relu(h @ W1 + b1)   [65536 x 512] x [512 x 1024]
    ffgemm_kernel<<<dim3(ROWS / 128, HID / 64), 256, 0, stream>>>(
        hb, D, w1b, HID, b1, nullptr, act1, nullptr, HID, D, 0);
    // GEMM2: hid = act1 @ W2 + b2 + h   [65536 x 1024] x [1024 x 512]
    ffgemm_kernel<<<dim3(ROWS / 128, D / 64), 256, 0, stream>>>(
        act1, HID, w2b, D, b2, hf, nullptr, hid, D, HID, 1);

    ln_kernel<<<ROWS / 8, 256, 0, stream>>>(hid, ln_g, ln_b, wgm, gate, (long)ROWS);
    topk_kernel<<<BB, 256, 0, stream>>>(gate, tidx);
    attn_kernel<<<BB, 256, 0, stream>>>(hid, tidx, Wr, br, ctx);
    outgemm_kernel<<<(VOCAB + 255) / 256, 256, 0, stream>>>(ctx, Wo, bo, out);
}
